// MultiHead_VLP_LSTM_52527450030160
// MI455X (gfx1250) — compile-verified, matrix operand order checked by reading, not run on hardware
//
#include <hip/hip_runtime.h>
#include <hip/hip_bf16.h>
#include <math.h>

// ---------------------------------------------------------------------------
// MultiHead VLP LSTM for MI455X (gfx1250, wave32, WMMA).
// Persistent-block design: 32 blocks x 256 threads (8 wave32s), each block
// owns 16 batch rows and walks all T=128 timesteps keeping the LSTM carry in
// LDS. LSTM gate GEMMs ([16x128]x[128x512] bf16) run on v_wmma_f32_16x16x32_bf16
// with weights pre-packed into B-fragment layout by a prep kernel. All small
// MLPs / attention / softmax / layernorm are scalar VALU, parallelized across
// the 256 threads. Problem is recurrence-latency bound (65 GFLOP total,
// ~1 MB of weights => L2 resident), so one launch + LDS-resident state is the
// right shape for this chip.
// ---------------------------------------------------------------------------

typedef __attribute__((ext_vector_type(16))) __bf16 v16bf;
typedef __attribute__((ext_vector_type(8)))  __bf16 v8bf;
typedef __attribute__((ext_vector_type(8)))  float  v8f;

#define Bsz   512
#define Tlen  128
#define RSS   12
#define NLED  36
#define FEAT  8
#define HD    64
#define LH    128
#define MB    16          // batch rows per block
#define NBLK  (Bsz/MB)    // 32 blocks
#define NTHR  256         // 8 wave32s

// ---- parameter pointer bundle (jax pytree order: dict keys sorted) --------
struct Params {
  const float *rss_seq, *init_pos, *led_pos_freq;
  const float *ctx_chimp_W, *ctx_chimp_b;
  const float *ctx_k0_W, *ctx_k0_b, *ctx_k1_W, *ctx_k1_b;
  const float *ctx_m0_W, *ctx_m0_b, *ctx_m1_W, *ctx_m1_b;
  const float *ctx_qf0_W, *ctx_qf0_b, *ctx_qf1_W, *ctx_qf1_b;
  const float *ctx_s2s_W, *ctx_s2s_b;
  const float *far_k0_W, *far_k0_b, *far_k1_W, *far_k1_b;
  const float *fq0_W, *fq0_b, *fq1_W, *fq1_b;
  const float *far_sigma;
  const float *fc0_W, *fc0_b, *fc1_W, *fc1_b;
  const float *fus0_W, *fus0_b, *fus1_W, *fus1_b;
  const float *le0_W, *le0_b, *le1_W, *le1_b;
  const float *ln_b, *ln_g;
  const float *l1_Whh, *l1_Wih, *l1_bhh, *l1_bih;
  const float *l2_Whh, *l2_Wih, *l2_bhh, *l2_bih;
  const float *ng0_W, *ng0_b, *ng1_W, *ng1_b;
  const float *nk0_W, *nk0_b, *nk1_W, *nk1_b;
  const float *nq0_W, *nq0_b, *nq1_W, *nq1_b;
  const float *near_sigma;
};

// ---- workspace layout -----------------------------------------------------
// f32 region: Kn|Kf|Kc (3*36*64), led_feat (36*8), led_pos (36*4)
#define WS_K     0
#define WS_FEAT  (WS_K + 3*NLED*HD)      // 6912
#define WS_POS   (WS_FEAT + NLED*FEAT)   // 7200
#define WS_FTOT  (WS_POS + NLED*4)       // 7344 floats (29376 B, 64B aligned)
// bf16 region: packed WMMA B fragments, 4 matrices x 4 ktiles x 32 ntiles,
// each fragment = 32 lanes x 16 bf16 (lane-contiguous).
#define PACK_FRAG   512                  // elements per fragment
#define PACK_MAT    (4*32*PACK_FRAG)     // 65536 elements per matrix
#define MAT_XIH1    0
#define MAT_HH1     (1*PACK_MAT)
#define MAT_XIH2    (2*PACK_MAT)
#define MAT_HH2     (3*PACK_MAT)
#define PACK_TOTAL  (4*PACK_MAT)         // 262144 bf16 = 512 KB

// ---- LDS layout (dynamic shared), float offsets ---------------------------
#define O_K      0                       // 3*36*64 = 6912
#define O_FEAT   (O_K + 3*NLED*HD)       // 288
#define O_POS    (O_FEAT + NLED*FEAT)    // 144
#define O_GATES  (O_POS + NLED*4)        // 16*512 = 8192
#define O_C1     (O_GATES + MB*512)      // 2048
#define O_C2     (O_C1 + MB*LH)
#define O_H2     (O_C2 + MB*LH)
#define O_MOT    (O_H2 + MB*LH)          // 16*32
#define O_MHID   (O_MOT + MB*32)         // 16*64
#define O_CW     (O_MHID + MB*64)        // 16*12
#define O_SIG2   (O_CW + MB*RSS)         // 16
#define O_HW     (O_SIG2 + MB)           // 16*4
#define O_PREV   (O_HW + MB*4)           // 16*4
#define O_FHID   (O_PREV + MB*4)         // 16*32
#define O_FCHID  (O_FHID + MB*32)        // 16*64
#define O_STATS  (O_FCHID + MB*64)       // 32
#define O_FEND   (O_STATS + 32)
#define SMEM_BYTES (O_FEND*4 + 3*MB*LH*2)   // + x_bf, h1_bf, h2_bf (bf16)

__device__ __forceinline__ float sigm(float x) { return 1.f / (1.f + __expf(-x)); }

// masked softmax over the 3 LEDs whose freq matches row r (n in {r,r+12,r+24});
// accumulates w * softmax(scores) @ led_feat into facc[8].
__device__ __forceinline__ void attn_acc(const float* Q, const float* Kb,
                                         const float* sPos, const float* sFeat,
                                         float px, float py, float pz,
                                         int r, float inv2s2, float w, float* facc) {
  float s[3]; int nn[3];
  #pragma unroll
  for (int j = 0; j < 3; ++j) {
    int n = r + 12 * j; nn[j] = n;
    const float* K = Kb + n * HD;
    float d = 0.f;
    #pragma unroll
    for (int k = 0; k < HD; ++k) d += Q[k] * K[k];
    float dx = px - sPos[n*4+0], dy = py - sPos[n*4+1], dz = pz - sPos[n*4+2];
    s[j] = d - (dx*dx + dy*dy + dz*dz) * inv2s2;
  }
  float mx = fmaxf(s[0], fmaxf(s[1], s[2]));
  float e0 = __expf(s[0]-mx), e1 = __expf(s[1]-mx), e2 = __expf(s[2]-mx);
  float inv = w / (e0 + e1 + e2);
  #pragma unroll
  for (int k = 0; k < FEAT; ++k)
    facc[k] += inv * (e0 * sFeat[nn[0]*FEAT+k] +
                      e1 * sFeat[nn[1]*FEAT+k] +
                      e2 * sFeat[nn[2]*FEAT+k]);
}

// ---------------------------------------------------------------------------
// Prep 1: time-invariant LED features and K projections -> workspace (f32)
// ---------------------------------------------------------------------------
__global__ void vlp_prep_led(Params P, float* __restrict__ wsf) {
  int n = threadIdx.x;
  if (n >= NLED) return;
  float lpf[4];
  #pragma unroll
  for (int i = 0; i < 4; ++i) lpf[i] = P.led_pos_freq[n*4 + i];
  // led encoder: 4 -> 32 relu -> 8 relu
  float h1[32];
  for (int j = 0; j < 32; ++j) {
    float a = P.le0_b[j];
    #pragma unroll
    for (int i = 0; i < 4; ++i) a += lpf[i] * P.le0_W[i*32 + j];
    h1[j] = fmaxf(a, 0.f);
  }
  float feat[FEAT];
  for (int d = 0; d < FEAT; ++d) {
    float a = P.le1_b[d];
    for (int j = 0; j < 32; ++j) a += h1[j] * P.le1_W[j*FEAT + d];
    feat[d] = fmaxf(a, 0.f);
    wsf[WS_FEAT + n*FEAT + d] = feat[d];
  }
  wsf[WS_POS + n*4 + 0] = lpf[0];
  wsf[WS_POS + n*4 + 1] = lpf[1];
  wsf[WS_POS + n*4 + 2] = lpf[2];
  wsf[WS_POS + n*4 + 3] = 0.f;
  float lin[FEAT + 3];
  #pragma unroll
  for (int d = 0; d < FEAT; ++d) lin[d] = feat[d];
  lin[FEAT+0] = lpf[0]; lin[FEAT+1] = lpf[1]; lin[FEAT+2] = lpf[2];
  const float* k0W[3] = { P.nk0_W, P.far_k0_W, P.ctx_k0_W };
  const float* k0b[3] = { P.nk0_b, P.far_k0_b, P.ctx_k0_b };
  const float* k1W[3] = { P.nk1_W, P.far_k1_W, P.ctx_k1_W };
  const float* k1b[3] = { P.nk1_b, P.far_k1_b, P.ctx_k1_b };
  for (int h = 0; h < 3; ++h) {
    float hh[32];
    for (int j = 0; j < 32; ++j) {
      float a = k0b[h][j];
      #pragma unroll
      for (int i = 0; i < FEAT+3; ++i) a += lin[i] * k0W[h][i*32 + j];
      hh[j] = fmaxf(a, 0.f);
    }
    for (int d = 0; d < HD; ++d) {
      float a = k1b[h][d];
      for (int j = 0; j < 32; ++j) a += hh[j] * k1W[h][j*HD + d];
      wsf[WS_K + h*NLED*HD + n*HD + d] = a;
    }
  }
}

// ---------------------------------------------------------------------------
// Prep 2: pack LSTM weights (f32 [K][512]) into bf16 WMMA B fragments.
// Fragment element (kt,nt,lane,e): K = kt*32 + (e>>3)*16 + (lane>>4)*8 + (e&7)
//                                  N = nt*16 + (lane&15)
// ---------------------------------------------------------------------------
__global__ void vlp_prep_pack(Params P, __bf16* __restrict__ packW) {
  const float* mats[4] = { P.l1_Wih, P.l1_Whh, P.l2_Wih, P.l2_Whh };
  const int kdim[4] = { RSS*(1+FEAT), LH, LH, LH };   // 108,128,128,128
  for (int p = blockIdx.x * blockDim.x + threadIdx.x; p < PACK_TOTAL;
       p += gridDim.x * blockDim.x) {
    int mat  = p >> 16;
    int rem  = p & 65535;
    int frag = rem >> 9;
    int lane = (rem >> 4) & 31;
    int e    = rem & 15;
    int kt = frag >> 5, nt = frag & 31;
    int K = kt*32 + ((e >> 3) << 4) + ((lane >> 4) << 3) + (e & 7);
    int N = (nt << 4) + (lane & 15);
    float v = (K < kdim[mat]) ? mats[mat][K*512 + N] : 0.f;
    packW[p] = (__bf16)v;
  }
}

// ---------------------------------------------------------------------------
// Main persistent kernel
// ---------------------------------------------------------------------------
__global__ void __launch_bounds__(NTHR, 2)
vlp_main(Params P, const __bf16* __restrict__ packW,
         const float* __restrict__ wsf, float* __restrict__ out) {
  extern __shared__ __align__(16) char smem[];
  float*  fb     = (float*)smem;
  float*  sK     = fb + O_K;
  float*  sFeat  = fb + O_FEAT;
  float*  sPos   = fb + O_POS;
  float*  sGates = fb + O_GATES;
  float*  sC1    = fb + O_C1;
  float*  sC2    = fb + O_C2;
  float*  sH2    = fb + O_H2;
  float*  sMot   = fb + O_MOT;
  float*  sMHid  = fb + O_MHID;
  float*  sCw    = fb + O_CW;
  float*  sSig2  = fb + O_SIG2;
  float*  sHw    = fb + O_HW;
  float*  sPrev  = fb + O_PREV;
  float*  sFHid  = fb + O_FHID;
  float*  sFcHid = fb + O_FCHID;
  float*  sStats = fb + O_STATS;
  __bf16* sXbf   = (__bf16*)(fb + O_FEND);
  __bf16* sH1bf  = sXbf  + MB*LH;
  __bf16* sH2bf  = sH1bf + MB*LH;

  const int tid = threadIdx.x;
  const int b0  = blockIdx.x * MB;

  // load time-invariant tables, zero carries
  for (int i = tid; i < 3*NLED*HD; i += NTHR) sK[i]    = wsf[WS_K + i];
  for (int i = tid; i < NLED*FEAT; i += NTHR) sFeat[i] = wsf[WS_FEAT + i];
  for (int i = tid; i < NLED*4;    i += NTHR) sPos[i]  = wsf[WS_POS + i];
  for (int i = tid; i < MB*LH; i += NTHR) {
    sC1[i] = 0.f; sC2[i] = 0.f; sH2[i] = 0.f;
    sXbf[i] = (__bf16)0.f; sH1bf[i] = (__bf16)0.f; sH2bf[i] = (__bf16)0.f;
  }
  if (tid < MB) {
    sPrev[tid*4+0] = P.init_pos[(b0+tid)*3+0];
    sPrev[tid*4+1] = P.init_pos[(b0+tid)*3+1];
    sPrev[tid*4+2] = P.init_pos[(b0+tid)*3+2];
    sPrev[tid*4+3] = 0.f;
  }
  __syncthreads();

  const float sn = P.near_sigma[0];
  const float sf = P.far_sigma[0];
  const float inv2n = 1.f / (2.f * sn * sn);
  const float inv2f = 1.f / (2.f * sf * sf);
  const int lane = tid & 31;
  const int wave = tid >> 5;
  const int am   = lane & 15;
  const int akh  = lane >> 4;

  for (int t = 0; t < Tlen; ++t) {
    // ---- phase 1: motion / chimp / s2s / fusion (use h2 from prev step) ----
    for (int it = tid; it < MB*64; it += NTHR) {
      int b = it >> 6, o = it & 63;
      const float* hr = sH2 + b*LH;
      float a = P.ctx_m0_b[o];
      for (int k = 0; k < LH; ++k) a += hr[k] * P.ctx_m0_W[k*64 + o];
      sMHid[it] = fmaxf(a, 0.f);
    }
    for (int it = tid; it < MB*32; it += NTHR) {
      int b = it >> 5, o = it & 31;
      const float* hr = sH2 + b*LH;
      float a = P.fus0_b[o];
      for (int k = 0; k < LH; ++k) a += hr[k] * P.fus0_W[k*32 + o];
      sFHid[it] = fmaxf(a, 0.f);
    }
    __syncthreads();
    for (int it = tid; it < MB*32; it += NTHR) {
      int b = it >> 5, o = it & 31;
      const float* hr = sMHid + b*64;
      float a = P.ctx_m1_b[o];
      for (int k = 0; k < 64; ++k) a += hr[k] * P.ctx_m1_W[k*32 + o];
      sMot[it] = a;                                  // linear output
    }
    __syncthreads();
    for (int it = tid; it < MB*RSS; it += NTHR) {
      int b = it / RSS, r = it - b*RSS;
      const float* mo = sMot + b*32;
      float a = P.ctx_chimp_b[r];
      for (int k = 0; k < 32; ++k) a += mo[k] * P.ctx_chimp_W[k*RSS + r];
      sCw[it] = sigm(a);
    }
    if (tid < MB) {
      int b = tid;
      const float* mo = sMot + b*32;
      float a = P.ctx_s2s_b[0];
      for (int k = 0; k < 32; ++k) a += mo[k] * P.ctx_s2s_W[k];
      float sc = log1pf(__expf(a)) + 0.5f;           // softplus + 0.5
      sSig2[b] = 1.f / (2.f * sc * sc);
      float l[3];
      #pragma unroll
      for (int o = 0; o < 3; ++o) {
        float acc = P.fus1_b[o];
        for (int k = 0; k < 32; ++k) acc += sFHid[b*32+k] * P.fus1_W[k*3 + o];
        l[o] = acc;
      }
      float mx = fmaxf(l[0], fmaxf(l[1], l[2]));
      float e0 = __expf(l[0]-mx), e1 = __expf(l[1]-mx), e2 = __expf(l[2]-mx);
      float inv = 1.f / (e0 + e1 + e2);
      sHw[b*4+0] = e0*inv; sHw[b*4+1] = e1*inv; sHw[b*4+2] = e2*inv;
    }
    __syncthreads();

    // ---- phase 2: per (b,r) gates + 3 attentions, stage x into bf16 -------
    if (tid < MB*RSS) {
      int b = tid / RSS, r = tid - b*RSS;
      float rss = P.rss_seq[((size_t)(b0+b)*Tlen + t)*RSS + r];
      float act = rss > 0.5f ? 1.f : 0.f;
      float ga = P.ng1_b[0];
      #pragma unroll
      for (int j = 0; j < 16; ++j) {
        float hj = fmaxf(P.ng0_W[j]*rss + P.ng0_b[j], 0.f);
        ga += hj * P.ng1_W[j];
      }
      float iw = act * sigm(ga);
      float px = sPrev[b*4], py = sPrev[b*4+1], pz = sPrev[b*4+2];
      float hw0 = sHw[b*4], hw1 = sHw[b*4+1], hw2 = sHw[b*4+2];
      float facc[FEAT] = {0,0,0,0,0,0,0,0};
      float Q[HD];
      // near: relu hidden
      float xin = rss * iw;
      #pragma unroll
      for (int d = 0; d < HD; ++d) Q[d] = P.nq1_b[d];
      for (int j = 0; j < 32; ++j) {
        float hj = fmaxf(P.nq0_W[j]*xin + P.nq0_b[j], 0.f);
        const float* w2 = P.nq1_W + j*HD;
        #pragma unroll
        for (int d = 0; d < HD; ++d) Q[d] += hj * w2[d];
      }
      attn_acc(Q, sK, sPos, sFeat, px, py, pz, r, inv2n, hw0, facc);
      // far: tanh hidden
      float weak = __expf(-fabsf(rss - 2.75f) * 0.5f) * act;
      xin = rss * weak;
      #pragma unroll
      for (int d = 0; d < HD; ++d) Q[d] = P.fq1_b[d];
      for (int j = 0; j < 32; ++j) {
        float hj = tanhf(P.fq0_W[j]*xin + P.fq0_b[j]);
        const float* w2 = P.fq1_W + j*HD;
        #pragma unroll
        for (int d = 0; d < HD; ++d) Q[d] += hj * w2[d];
      }
      attn_acc(Q, sK + NLED*HD, sPos, sFeat, px, py, pz, r, inv2f, hw1, facc);
      // ctx: [rss, motion(32)] -> 64 relu -> 64, scaled by cw
      const float* mo = sMot + b*32;
      #pragma unroll
      for (int d = 0; d < HD; ++d) Q[d] = P.ctx_qf1_b[d];
      for (int j = 0; j < 64; ++j) {
        float hj = P.ctx_qf0_b[j] + P.ctx_qf0_W[j] * rss;
        for (int k = 0; k < 32; ++k) hj += mo[k] * P.ctx_qf0_W[(1+k)*64 + j];
        hj = fmaxf(hj, 0.f);
        const float* w2 = P.ctx_qf1_W + j*HD;
        #pragma unroll
        for (int d = 0; d < HD; ++d) Q[d] += hj * w2[d];
      }
      float cw = sCw[b*RSS + r];
      #pragma unroll
      for (int d = 0; d < HD; ++d) Q[d] *= cw;
      attn_acc(Q, sK + 2*NLED*HD, sPos, sFeat, px, py, pz, r, sSig2[b], hw2, facc);
      // stage x row: x[r*9] = rss, x[r*9+1..8] = fused (hw-weighted sum)
      __bf16* xr = sXbf + b*LH + r*(1+FEAT);
      xr[0] = (__bf16)rss;
      #pragma unroll
      for (int k = 0; k < FEAT; ++k) xr[1+k] = (__bf16)facc[k];
    }
    __syncthreads();

    // ---- phase 3: LSTM1 gates via WMMA (x@Wih1 + h1@Whh1) -----------------
    {
      #pragma unroll 1
      for (int nt = wave*4; nt < wave*4 + 4; ++nt) {
        v8f acc = {0.f,0.f,0.f,0.f,0.f,0.f,0.f,0.f};
        union { v16bf v; v8bf h[2]; } a;
        #pragma unroll
        for (int kt = 0; kt < 4; ++kt) {
          const __bf16* ab = sXbf + am*LH + kt*32 + akh*8;
          a.h[0] = *(const v8bf*)ab;
          a.h[1] = *(const v8bf*)(ab + 16);
          const __bf16* bp = packW + MAT_XIH1 + ((kt*32 + nt) << 9) + (lane << 4);
          __builtin_prefetch((const void*)(bp + PACK_FRAG), 0, 1);
          v16bf bm = *(const v16bf*)bp;
          acc = __builtin_amdgcn_wmma_f32_16x16x32_bf16(false, a.v, false, bm,
                                                        (short)0, acc, false, false);
        }
        #pragma unroll
        for (int kt = 0; kt < 4; ++kt) {
          const __bf16* ab = sH1bf + am*LH + kt*32 + akh*8;
          a.h[0] = *(const v8bf*)ab;
          a.h[1] = *(const v8bf*)(ab + 16);
          const __bf16* bp = packW + MAT_HH1 + ((kt*32 + nt) << 9) + (lane << 4);
          __builtin_prefetch((const void*)(bp + PACK_FRAG), 0, 1);
          v16bf bm = *(const v16bf*)bp;
          acc = __builtin_amdgcn_wmma_f32_16x16x32_bf16(false, a.v, false, bm,
                                                        (short)0, acc, false, false);
        }
        #pragma unroll
        for (int j = 0; j < 8; ++j)                    // C layout: row j(+8), col lane&15
          sGates[(j + akh*8)*512 + nt*16 + am] = acc[j];
      }
    }
    __syncthreads();

    // ---- phase 4: LSTM1 cell update ---------------------------------------
    for (int it = tid; it < MB*LH; it += NTHR) {
      int b = it >> 7, h = it & 127;
      const float* g = sGates + b*512;
      float gi = g[h]       + P.l1_bih[h]       + P.l1_bhh[h];
      float gf = g[128 + h] + P.l1_bih[128 + h] + P.l1_bhh[128 + h];
      float gg = g[256 + h] + P.l1_bih[256 + h] + P.l1_bhh[256 + h];
      float go = g[384 + h] + P.l1_bih[384 + h] + P.l1_bhh[384 + h];
      float c = sigm(gf) * sC1[it] + sigm(gi) * tanhf(gg);
      float hn = sigm(go) * tanhf(c);
      sC1[it] = c;
      sH1bf[it] = (__bf16)hn;
    }
    __syncthreads();

    // ---- phase 5: LSTM2 gates via WMMA (h1@Wih2 + h2@Whh2) ----------------
    {
      #pragma unroll 1
      for (int nt = wave*4; nt < wave*4 + 4; ++nt) {
        v8f acc = {0.f,0.f,0.f,0.f,0.f,0.f,0.f,0.f};
        union { v16bf v; v8bf h[2]; } a;
        #pragma unroll
        for (int kt = 0; kt < 4; ++kt) {
          const __bf16* ab = sH1bf + am*LH + kt*32 + akh*8;
          a.h[0] = *(const v8bf*)ab;
          a.h[1] = *(const v8bf*)(ab + 16);
          const __bf16* bp = packW + MAT_XIH2 + ((kt*32 + nt) << 9) + (lane << 4);
          __builtin_prefetch((const void*)(bp + PACK_FRAG), 0, 1);
          v16bf bm = *(const v16bf*)bp;
          acc = __builtin_amdgcn_wmma_f32_16x16x32_bf16(false, a.v, false, bm,
                                                        (short)0, acc, false, false);
        }
        #pragma unroll
        for (int kt = 0; kt < 4; ++kt) {
          const __bf16* ab = sH2bf + am*LH + kt*32 + akh*8;
          a.h[0] = *(const v8bf*)ab;
          a.h[1] = *(const v8bf*)(ab + 16);
          const __bf16* bp = packW + MAT_HH2 + ((kt*32 + nt) << 9) + (lane << 4);
          __builtin_prefetch((const void*)(bp + PACK_FRAG), 0, 1);
          v16bf bm = *(const v16bf*)bp;
          acc = __builtin_amdgcn_wmma_f32_16x16x32_bf16(false, a.v, false, bm,
                                                        (short)0, acc, false, false);
        }
        #pragma unroll
        for (int j = 0; j < 8; ++j)
          sGates[(j + akh*8)*512 + nt*16 + am] = acc[j];
      }
    }
    __syncthreads();

    // ---- phase 6: LSTM2 cell update ---------------------------------------
    for (int it = tid; it < MB*LH; it += NTHR) {
      int b = it >> 7, h = it & 127;
      const float* g = sGates + b*512;
      float gi = g[h]       + P.l2_bih[h]       + P.l2_bhh[h];
      float gf = g[128 + h] + P.l2_bih[128 + h] + P.l2_bhh[128 + h];
      float gg = g[256 + h] + P.l2_bih[256 + h] + P.l2_bhh[256 + h];
      float go = g[384 + h] + P.l2_bih[384 + h] + P.l2_bhh[384 + h];
      float c = sigm(gf) * sC2[it] + sigm(gi) * tanhf(gg);
      float hn = sigm(go) * tanhf(c);
      sC2[it] = c;
      sH2[it] = hn;
      sH2bf[it] = (__bf16)hn;
    }
    __syncthreads();

    // ---- phase 7: layernorm + fc head -------------------------------------
    if (tid < MB) {
      const float* hr = sH2 + tid*LH;
      float s = 0.f, s2 = 0.f;
      for (int k = 0; k < LH; ++k) { float v = hr[k]; s += v; s2 += v*v; }
      float mean = s * (1.f/LH);
      float var  = s2 * (1.f/LH) - mean*mean;
      sStats[tid]      = mean;
      sStats[MB + tid] = rsqrtf(var + 1e-5f);
    }
    __syncthreads();
    float* sLn = sGates;                               // reuse gate buffer
    for (int it = tid; it < MB*LH; it += NTHR) {
      int b = it >> 7, k = it & 127;
      float v = (sH2[it] - sStats[b]) * sStats[MB + b];
      sLn[it] = v * P.ln_g[k] + P.ln_b[k];
    }
    __syncthreads();
    for (int it = tid; it < MB*64; it += NTHR) {
      int b = it >> 6, o = it & 63;
      const float* lr = sLn + b*LH;
      float a = P.fc0_b[o];
      for (int k = 0; k < LH; ++k) a += lr[k] * P.fc0_W[k*64 + o];
      sFcHid[it] = fmaxf(a, 0.f);
    }
    __syncthreads();
    if (tid < MB) {
      int b = tid;
      const float* hr = sFcHid + b*64;
      #pragma unroll
      for (int o = 0; o < 3; ++o) {
        float a = P.fc1_b[o];
        for (int k = 0; k < 64; ++k) a += hr[k] * P.fc1_W[k*3 + o];
        sPrev[b*4 + o] = a;                            // prev_pos for next step
        out[((size_t)(b0+b)*Tlen + t)*3 + o] = a;
      }
    }
    __syncthreads();
  }
}

// ---------------------------------------------------------------------------
// Host launcher. Input order assumption (compile-only, documented):
// top-level dict insertion order {rss_seq, init_pos, led_pos_freq, params},
// params flattened as a jax pytree (dict keys sorted alphabetically, lists
// in order, each lp as {'W','b'} and lstm_p as {'Whh','Wih','bhh','bih'}).
// ---------------------------------------------------------------------------
extern "C" void kernel_launch(void* const* d_in, const int* in_sizes, int n_in,
                              void* d_out, int out_size, void* d_ws, size_t ws_size,
                              hipStream_t stream) {
  const float* const* in = (const float* const*)d_in;
  Params P;
  P.rss_seq = in[0]; P.init_pos = in[1]; P.led_pos_freq = in[2];
  // params pytree leaves (sorted keys): ctx, far, fc, fusion, led_enc, ln, lstm, near
  P.ctx_chimp_W = in[3];  P.ctx_chimp_b = in[4];
  P.ctx_k0_W = in[5];  P.ctx_k0_b = in[6];  P.ctx_k1_W = in[7];  P.ctx_k1_b = in[8];
  P.ctx_m0_W = in[9];  P.ctx_m0_b = in[10]; P.ctx_m1_W = in[11]; P.ctx_m1_b = in[12];
  P.ctx_qf0_W = in[13]; P.ctx_qf0_b = in[14]; P.ctx_qf1_W = in[15]; P.ctx_qf1_b = in[16];
  P.ctx_s2s_W = in[17]; P.ctx_s2s_b = in[18];
  P.far_k0_W = in[19]; P.far_k0_b = in[20]; P.far_k1_W = in[21]; P.far_k1_b = in[22];
  P.fq0_W = in[23]; P.fq0_b = in[24]; P.fq1_W = in[25]; P.fq1_b = in[26];
  P.far_sigma = in[27];
  P.fc0_W = in[28]; P.fc0_b = in[29]; P.fc1_W = in[30]; P.fc1_b = in[31];
  P.fus0_W = in[32]; P.fus0_b = in[33]; P.fus1_W = in[34]; P.fus1_b = in[35];
  P.le0_W = in[36]; P.le0_b = in[37]; P.le1_W = in[38]; P.le1_b = in[39];
  P.ln_b = in[40]; P.ln_g = in[41];
  P.l1_Whh = in[42]; P.l1_Wih = in[43]; P.l1_bhh = in[44]; P.l1_bih = in[45];
  P.l2_Whh = in[46]; P.l2_Wih = in[47]; P.l2_bhh = in[48]; P.l2_bih = in[49];
  P.ng0_W = in[50]; P.ng0_b = in[51]; P.ng1_W = in[52]; P.ng1_b = in[53];
  P.nk0_W = in[54]; P.nk0_b = in[55]; P.nk1_W = in[56]; P.nk1_b = in[57];
  P.nq0_W = in[58]; P.nq0_b = in[59]; P.nq1_W = in[60]; P.nq1_b = in[61];
  P.near_sigma = in[62];

  float*  wsf   = (float*)d_ws;
  __bf16* packW = (__bf16*)((char*)d_ws + WS_FTOT * sizeof(float));

  vlp_prep_led<<<1, 64, 0, stream>>>(P, wsf);
  vlp_prep_pack<<<256, 256, 0, stream>>>(P, packW);
  vlp_main<<<NBLK, NTHR, SMEM_BYTES, stream>>>(P, packW, wsf, (float*)d_out);
}
